// _SS2D_spatial_61452392071736
// MI455X (gfx1250) — matrix operand order repaired and checked by reading, not compiled, not run on hardware
//
#include <hip/hip_runtime.h>
#include <hip/hip_bf16.h>

// ---------------- problem constants ----------------
#define BSZ   2
#define HH    64
#define WW    64
#define LL    4096          // H*W
#define BL    8192          // B*L
#define DM    128           // d_model
#define DI    256           // d_inner
#define NST   16            // d_state
#define RNK   8             // dt_rank
#define KD    4             // directions
#define CPAD  48            // padded (dt_rank + 2*d_state) = 40 -> 48

typedef __bf16 bf16;
typedef __attribute__((ext_vector_type(16))) __bf16 v16bf;
typedef __attribute__((ext_vector_type(8)))  float  v8f;

union FragU { unsigned int u[8]; v16bf v; };

#if defined(__has_builtin)
#if __has_builtin(__builtin_amdgcn_tensor_load_to_lds) && __has_builtin(__builtin_amdgcn_s_wait_tensorcnt)
#define HAVE_TDM 1
#endif
#endif

#ifdef HAVE_TDM
#if __clang_major__ >= 23
#warning "CDNA5 probe: TDM builtin ENABLED (6-arg form)"
#else
#warning "CDNA5 probe: TDM builtin ENABLED (5-arg form)"
#endif
#else
#warning "CDNA5 probe: TDM builtin NOT available; cooperative staging only"
#endif

#ifdef HAVE_TDM
typedef unsigned int u32x4 __attribute__((ext_vector_type(4)));
typedef int          i32x4 __attribute__((ext_vector_type(4)));
typedef int          i32x8 __attribute__((ext_vector_type(8)));

// 2-D TDM load: rows x dim0 bf16 elements, row stride stride0 (elements) -> LDS
__device__ __forceinline__ void tdm_load_2d(const void* gptr, unsigned lds_off,
                                            unsigned dim0, unsigned rows,
                                            unsigned stride0) {
  unsigned long long ga = (unsigned long long)gptr;
  u32x4 g0;
  g0[0] = 1u;                                               // count=1, load, no gather
  g0[1] = lds_off;                                          // LDS byte address
  g0[2] = (unsigned)ga;                                     // global_addr[31:0]
  g0[3] = (unsigned)((ga >> 32) & 0x01FFFFFFu) | (2u << 30);// [56:32] | type=2
  i32x8 g1;
  g1[0] = (int)(1u << 16);                                  // data_size=1 -> 2 bytes
  g1[1] = (int)((dim0 & 0xFFFFu) << 16);                    // tensor_dim0[15:0]
  g1[2] = (int)((dim0 >> 16) | ((rows & 0xFFFFu) << 16));   // dim0 hi | tensor_dim1 lo
  g1[3] = (int)((rows >> 16) | (dim0 << 16));               // dim1 hi | tile_dim0
  g1[4] = (int)rows;                                        // tile_dim1, tile_dim2=0
  g1[5] = (int)stride0;                                     // tensor_dim0_stride lo
  g1[6] = 0;
  g1[7] = 0;                                                // dim1 stride unused (2D)
  i32x4 z4 = {0, 0, 0, 0};
#if __clang_major__ >= 23
  i32x8 z8 = {};
  __builtin_amdgcn_tensor_load_to_lds(g0, g1, z4, z4, z8, 0);
#else
  __builtin_amdgcn_tensor_load_to_lds(g0, g1, z4, z4, 0);
#endif
}
#endif // HAVE_TDM

// scan index j  ->  spatial index l (HW row-major) for direction k
__device__ __forceinline__ int lmap(int k, int j) {
  switch (k & 3) {
    case 0: return j;
    case 1: return ((j & 63) << 6) | (j >> 6);          // WH order (H==W==64)
    case 2: return (LL - 1) - j;                        // reversed HW
    default: { int t = (LL - 1) - j; return ((t & 63) << 6) | (t >> 6); }
  }
}

__device__ __forceinline__ float sigmoidf_(float x) { return 1.f / (1.f + __expf(-x)); }

// ---------------- elementwise helpers ----------------
__global__ void k_cast_bf16(const float* __restrict__ s, bf16* __restrict__ d, int n) {
  int i = blockIdx.x * 256 + threadIdx.x;
  if (i < n) d[i] = (bf16)s[i];
}

// x_proj_weight (4,40,256) fp32 -> (4,48,256) bf16, zero-padded rows 40..47
__global__ void k_pack_xpw(const float* __restrict__ s, bf16* __restrict__ d) {
  int i = blockIdx.x * 256 + threadIdx.x;
  if (i >= KD * CPAD * DI) return;
  int k  = i / (CPAD * DI);
  int r  = i % (CPAD * DI);
  int c  = r / DI;
  int dd = r % DI;
  d[i] = (c < RNK + 2 * NST) ? (bf16)s[(k * (RNK + 2 * NST) + c) * DI + dd] : (bf16)0.f;
}

// ---------------- WMMA fragment loaders ----------------
__device__ __forceinline__ void load_afrag(FragU& f, const bf16* sArow, int k0, int half) {
#pragma unroll
  for (int v = 0; v < 8; ++v) {
    // A (16x32, 16-bit): VGPR v -> K = (v/4)*16 + half*8 + (v%4)*2  (+k0)
    const int ka = k0 + ((v >> 2) << 4) + (half << 3) + ((v & 3) << 1);
    f.u[v] = *(const unsigned int*)(sArow + ka);
  }
}
__device__ __forceinline__ void load_bfrag(FragU& f, const bf16* Wrow, int k0, int half) {
#pragma unroll
  for (int v = 0; v < 8; ++v) {
    // B (32x16, 16-bit): VGPR v -> K = (v/4)*16 + half*8 + (v%4)*2  (+k0), same as A
    f.u[v] = *(const unsigned int*)(Wrow + k0 + ((v >> 2) << 4) + (half << 3) + ((v & 3) << 1));
  }
}

// ---------------- WMMA GEMM:  C(MxN) = A(MxK) * W(NxK)^T  ----------------
// KK is compile-time (128 or 256) -> fully unrolled K chain, all fragments
// preloaded into registers ahead of the WMMA chain (partial waits, no stalls).
// Block owns one 16-row m-panel of A (LDS; TDM when describable) and
// blockDim/32 consecutive n-tiles.  grid = (ceil(N/16/wpb), M/16, batches).
// XPROJ: blockIdx.z = b*4+k ; A rows spatially remapped via lmap().
template <int KK, int XPROJ>
__global__ void k_wmma_gemm(const bf16* __restrict__ A, const bf16* __restrict__ W,
                            float* __restrict__ C, int M, int N, int ldc) {
  constexpr int NS = KK / 32;                     // K steps
  __shared__ alignas(16) bf16 sA[16 * KK];        // 16 x K panel
  const int wave = threadIdx.x >> 5;
  const int lane = threadIdx.x & 31;
  const int m0 = blockIdx.y << 4;

  int kdir = -1;
  if (XPROJ) {
    const int y = blockIdx.z;
    const int b = y >> 2;
    kdir = y & 3;
    A += (long)b * LL * DI;
    W += (long)kdir * CPAD * DI;
    C += (long)y * LL * CPAD;
  }

  // ---- stage the 16 x K A-panel into LDS (shared by all waves of the block)
#ifdef HAVE_TDM
  if (!XPROJ) {
    if (wave == 0) {
      if (lane == 0) {
        tdm_load_2d(A + (long)m0 * KK, (unsigned)(size_t)sA, (unsigned)KK, 16u,
                    (unsigned)KK);
      }
      __builtin_amdgcn_s_wait_tensorcnt(0);
    }
  } else
#endif
  {
    for (int e = threadIdx.x * 8; e < 16 * KK; e += blockDim.x * 8) {
      const int r = e / KK, col = e % KK;
      const int arow = (kdir < 0) ? (m0 + r) : lmap(kdir, m0 + r);
      *(float4*)(&sA[e]) = *(const float4*)(A + (long)arow * KK + col);
    }
  }
  __syncthreads();

  const int nt = blockIdx.x * (blockDim.x >> 5) + wave;
  const int n0 = nt << 4;
  if (n0 >= N) return;                            // wave-uniform

  const int half = lane >> 4;                     // 0: lanes 0-15, 1: lanes 16-31
  const int mr   = lane & 15;
  const bf16* Wrow  = W + (long)(n0 + mr) * KK;   // lane's N index == mr for B frag
  const bf16* sArow = sA + mr * KK;               // lane's M row for A frag

  // preload all fragments, then run the WMMA chain (loads overlap matrix ops)
  FragU a[NS], b[NS];
#pragma unroll
  for (int s = 0; s < NS; ++s) {
    load_bfrag(b[s], Wrow, s * 32, half);
    load_afrag(a[s], sArow, s * 32, half);
  }
  v8f acc = {};
#pragma unroll
  for (int s = 0; s < NS; ++s) {
    acc = __builtin_amdgcn_wmma_f32_16x16x32_bf16(
        false, a[s].v, false, b[s].v, (short)0, acc, false, false);
  }
#pragma unroll
  for (int r = 0; r < 8; ++r) {                   // C: VGPR r -> rows r / r+8
    const int m = m0 + r + (half << 3);
    C[(long)m * ldc + n0 + mr] = acc[r];
  }
}

// ---------------- depthwise 3x3 conv + bias + SiLU ----------------
// in : xz (BL x 512), cols 0..255  ; out: xc fp32 (B,L,256) and xcb bf16
__global__ void k_conv_silu(const float* __restrict__ xz, const float* __restrict__ wc,
                            const float* __restrict__ bc,
                            float* __restrict__ xc, bf16* __restrict__ xcb) {
  const int pos = blockIdx.x;                   // h*W + w
  const int b   = blockIdx.y;
  const int c   = threadIdx.x;
  const int h = pos >> 6, w = pos & 63;
  float acc = bc[c];
#pragma unroll
  for (int dh = -1; dh <= 1; ++dh) {
    const int hh = h + dh;
    if (hh < 0 || hh >= HH) continue;
#pragma unroll
    for (int dw = -1; dw <= 1; ++dw) {
      const int ww2 = w + dw;
      if (ww2 < 0 || ww2 >= WW) continue;
      acc += xz[((long)(b * LL + hh * WW + ww2)) * (2 * DI) + c] *
             wc[c * 9 + (dh + 1) * 3 + (dw + 1)];
    }
  }
  const float s = acc * sigmoidf_(acc);
  const long o = (long)(b * LL + pos) * DI + c;
  xc[o]  = s;
  xcb[o] = (bf16)s;
}

// ---------------- selective scan (one block per (b,k), one lane per d) ----------
// Double-buffered x_dbl row, prefetched with CDNA5 async global->LDS loads
// (ASYNCcnt-tracked, no VGPR round-trip); one barrier per step.
__global__ void k_scan(const float* __restrict__ xc,      // (B,L,256) spatial
                       const float* __restrict__ xdbl,    // (B*K, L, 48) scan order
                       const float* __restrict__ dtw,     // (K,256,8)
                       const float* __restrict__ dtb,     // (K,256)
                       const float* __restrict__ Alog,    // (K*256,16)
                       const float* __restrict__ Dsv,     // (K*256)
                       float* __restrict__ ym) {          // (B,L,256) accum
  const int bk = blockIdx.x;
  const int b = bk >> 2, k = bk & 3;
  const int d = threadIdx.x;
  const int kd = k * DI + d;

  float Aa[NST], h[NST], wdt[RNK];
#pragma unroll
  for (int n = 0; n < NST; ++n) { Aa[n] = -__expf(Alog[kd * NST + n]); h[n] = 0.f; }
#pragma unroll
  for (int r = 0; r < RNK; ++r) wdt[r] = dtw[kd * RNK + r];
  const float bias = dtb[kd];
  const float Dd   = Dsv[kd];

  __shared__ float srow[2][CPAD];
  const float* drow = xdbl + (long)bk * LL * CPAD;
  const unsigned long long dbase = (unsigned long long)drow;
  if (d < CPAD) srow[0][d] = drow[d];
  __syncthreads();

  for (int j = 0; j < LL; ++j) {
    const float* cur = srow[j & 1];
    if (j + 1 < LL && d < CPAD) {                 // async prefetch next row
      const unsigned lds_a = (unsigned)(size_t)&srow[(j + 1) & 1][d];
      const int voff = ((j + 1) * CPAD + d) * 4;
      asm volatile("global_load_async_to_lds_b32 %0, %1, %2"
                   :: "v"(lds_a), "v"(voff), "s"(dbase) : "memory");
    }

    const int l = lmap(k, j);
    const float u = xc[(long)(b * LL + l) * DI + d];
    if (j + 1 < LL)
      __builtin_prefetch(&xc[(long)(b * LL + lmap(k, j + 1)) * DI + d], 0, 0);

    float dtr = bias;
#pragma unroll
    for (int r = 0; r < RNK; ++r) dtr += cur[r] * wdt[r];
    const float delta = (dtr > 20.f) ? dtr : log1pf(__expf(dtr));   // softplus
    const float du = delta * u;

    float y = 0.f;
#pragma unroll
    for (int n = 0; n < NST; ++n) {
      h[n] = __expf(delta * Aa[n]) * h[n] + du * cur[RNK + n];
      y += h[n] * cur[RNK + NST + n];
    }
    y += u * Dd;
    atomicAdd(&ym[(long)(b * LL + l) * DI + d], y);   // merge 4 directions

    asm volatile("s_wait_asynccnt 0x0" ::: "memory");
    __syncthreads();
  }
}

// ---------------- LayerNorm(256) * silu(z) -> bf16 ----------------
__global__ void k_ln_gate(const float* __restrict__ ym, const float* __restrict__ xz,
                          const float* __restrict__ g, const float* __restrict__ be,
                          bf16* __restrict__ out) {
  const int bl = blockIdx.x;
  const int c  = threadIdx.x;
  const float v = ym[(long)bl * DI + c];
  __shared__ float s1[DI], s2[DI];
  s1[c] = v; s2[c] = v * v;
  __syncthreads();
  for (int off = DI / 2; off > 0; off >>= 1) {
    if (c < off) { s1[c] += s1[c + off]; s2[c] += s2[c + off]; }
    __syncthreads();
  }
  const float mu  = s1[0] * (1.f / DI);
  const float var = s2[0] * (1.f / DI) - mu * mu;
  const float yn  = (v - mu) * rsqrtf(var + 1e-5f) * g[c] + be[c];
  const float zv  = xz[(long)bl * (2 * DI) + DI + c];
  out[(long)bl * DI + c] = (bf16)(yn * (zv * sigmoidf_(zv)));
}

// ---------------- host launch ----------------
extern "C" void kernel_launch(void* const* d_in, const int* in_sizes, int n_in,
                              void* d_out, int out_size, void* d_ws, size_t ws_size,
                              hipStream_t stream) {
  (void)in_sizes; (void)n_in; (void)out_size; (void)ws_size;
  const float* x    = (const float*)d_in[0];
  const float* Win  = (const float*)d_in[1];
  const float* Wcv  = (const float*)d_in[2];
  const float* bcv  = (const float*)d_in[3];
  const float* xpw  = (const float*)d_in[4];
  const float* dtw  = (const float*)d_in[5];
  const float* dtb  = (const float*)d_in[6];
  const float* Alog = (const float*)d_in[7];
  const float* Dsv  = (const float*)d_in[8];
  const float* lng  = (const float*)d_in[9];
  const float* lnb  = (const float*)d_in[10];
  const float* Wout = (const float*)d_in[11];
  float* out = (float*)d_out;

  char* ws = (char*)d_ws;
  size_t off = 0;
  auto alloc = [&](size_t bytes) { void* p = ws + off; off = (off + bytes + 255) & ~(size_t)255; return p; };
  float* xz    = (float*)alloc((size_t)BL * 2 * DI * 4);     // 16 MiB
  float* xc    = (float*)alloc((size_t)BL * DI * 4);         //  8 MiB
  float* ym    = (float*)alloc((size_t)BL * DI * 4);         //  8 MiB
  float* xdbl  = (float*)alloc((size_t)BSZ * KD * LL * CPAD * 4); // 6 MiB
  bf16* xb     = (bf16*)alloc((size_t)BL * DM * 2);
  bf16* xcb    = (bf16*)alloc((size_t)BL * DI * 2);
  bf16* ybg    = (bf16*)alloc((size_t)BL * DI * 2);
  bf16* Winb   = (bf16*)alloc((size_t)2 * DI * DM * 2);
  bf16* Woutb  = (bf16*)alloc((size_t)DM * DI * 2);
  bf16* xpwb   = (bf16*)alloc((size_t)KD * CPAD * DI * 2);

  // bf16 staging
  k_cast_bf16<<<(BL * DM + 255) / 256, 256, 0, stream>>>(x, xb, BL * DM);
  k_cast_bf16<<<(2 * DI * DM + 255) / 256, 256, 0, stream>>>(Win, Winb, 2 * DI * DM);
  k_cast_bf16<<<(DM * DI + 255) / 256, 256, 0, stream>>>(Wout, Woutb, DM * DI);
  k_pack_xpw<<<(KD * CPAD * DI + 255) / 256, 256, 0, stream>>>(xpw, xpwb);
  hipMemsetAsync(ym, 0, (size_t)BL * DI * 4, stream);

  // in_proj: (8192 x 128) @ (512 x 128)^T -> xz (8192 x 512)
  k_wmma_gemm<DM, 0><<<dim3(4, BL / 16, 1), 8 * 32, 0, stream>>>(
      xb, Winb, xz, BL, 2 * DI, 2 * DI);
  // depthwise conv + SiLU
  k_conv_silu<<<dim3(LL, BSZ), DI, 0, stream>>>(xz, Wcv, bcv, xc, xcb);
  // x_proj per (b,k): (4096 x 256 via lmap) @ (48 x 256)^T -> xdbl  (3 n-tiles)
  k_wmma_gemm<DI, 1><<<dim3(1, LL / 16, BSZ * KD), 3 * 32, 0, stream>>>(
      xcb, xpwb, xdbl, LL, CPAD, CPAD);
  // selective scan + directional merge
  k_scan<<<BSZ * KD, DI, 0, stream>>>(xc, xdbl, dtw, dtb, Alog, Dsv, ym);
  // LayerNorm + silu gate
  k_ln_gate<<<BL, DI, 0, stream>>>(ym, xz, lng, lnb, ybg);
  // out_proj: (8192 x 256) @ (128 x 256)^T -> out (8192 x 128)
  k_wmma_gemm<DI, 0><<<dim3(1, BL / 16, 1), 8 * 32, 0, stream>>>(
      ybg, Woutb, out, BL, DM, DM);
}
